// ProteinGAT_28355374088745
// MI455X (gfx1250) — matrix operand order repaired and checked by reading, compiled, not run on hardware
//
#include <hip/hip_runtime.h>
#include <cfloat>
#include <cmath>

// ---------------------------------------------------------------------------
// ProteinGAT (3-layer GATConv) for MI455X / gfx1250.
// Dense node transforms use v_wmma_f32_16x16x32_bf16 (wave32 WMMA, f32 acc).
// Edge softmax/aggregation is HBM/atomic bound (~1.4GB moved @23.3TB/s) and
// uses plain VALU kernels with f32 atomics.
// ---------------------------------------------------------------------------

typedef __bf16 bf16;
typedef __attribute__((ext_vector_type(16))) __bf16 v16bf;
typedef __attribute__((ext_vector_type(8)))  __bf16 v8bf;
typedef __attribute__((ext_vector_type(8)))  float  v8f;

#define TPB 256

static inline unsigned cdiv_u(size_t a, size_t b) { return (unsigned)((a + b - 1) / b); }

// ---------------- small utility kernels ----------------

__global__ void fill_f32_kernel(float* __restrict__ p, float v, size_t n) {
  size_t i = (size_t)blockIdx.x * blockDim.x + threadIdx.x;
  if (i < n) p[i] = v;
}

__global__ void f32_to_bf16_kernel(const float* __restrict__ s, bf16* __restrict__ d, size_t n) {
  size_t i = (size_t)blockIdx.x * blockDim.x + threadIdx.x;
  if (i < n) d[i] = (bf16)s[i];
}

// ---------------- WMMA GEMM: C[M,N] = A[M,K] * B[K,N] (bf16 in, f32 out) ----
// One wave computes a 16x64 output strip: 4 accumulators, A fragment reused 4x.
// Fragment layouts per CDNA5 ISA 7.12.2 (wave32):
//   A 16x32: lane&15 = M; half=lane>>4; elems 0..7 -> K=k0+half*8+j,
//            elems 8..15 -> K=k0+16+half*8+j
//   B 32x16: lane&15 = N; elems use the same K order as A
//   C/D    : vgpr r -> M = r + 8*half; col = lane&15

__global__ void wmma_gemm_bf16_kernel(const bf16* __restrict__ A,
                                      const bf16* __restrict__ B,
                                      float* __restrict__ C,
                                      int M, int N, int K) {
  const int lane = threadIdx.x & 31;
  const int wave = blockIdx.x * (blockDim.x >> 5) + (threadIdx.x >> 5);
  const int ng4 = N >> 6;        // number of 64-wide column groups
  const int mtiles = M >> 4;
  if (wave >= mtiles * ng4) return;          // wave-uniform exit (EXEC all-1 inside)
  const int mt = wave / ng4;
  const int ng = wave - mt * ng4;
  const int half = lane >> 4;
  const int col  = lane & 15;
  const int row  = (mt << 4) + col;
  const int n0   = ng << 6;

  v8f acc0 = {}, acc1 = {}, acc2 = {}, acc3 = {};

  for (int k0 = 0; k0 < K; k0 += 32) {
    // A fragment: two 16B loads, interleave into v16bf
    const bf16* ap = A + (size_t)row * K + k0 + half * 8;
    v8bf alo = *(const v8bf*)(ap);
    v8bf ahi = *(const v8bf*)(ap + 16);
    v16bf afrag = __builtin_shufflevector(alo, ahi,
        0, 1, 2, 3, 4, 5, 6, 7, 8, 9, 10, 11, 12, 13, 14, 15);

    // B fragments: lane&15 -> N col; elems follow A's K order; 4 consecutive 16-col tiles
    const bf16* bp = B + (size_t)(k0 + half * 8) * N + n0 + col;
    v16bf b0, b1, b2, b3;
#pragma unroll
    for (int s = 0; s < 16; ++s) {
      const size_t ko = (size_t)((s < 8) ? s : s + 8) * N;
      b0[s] = bp[ko];
      b1[s] = bp[ko + 16];
      b2[s] = bp[ko + 32];
      b3[s] = bp[ko + 48];
    }

    acc0 = __builtin_amdgcn_wmma_f32_16x16x32_bf16(false, afrag, false, b0, (short)0, acc0, false, false);
    acc1 = __builtin_amdgcn_wmma_f32_16x16x32_bf16(false, afrag, false, b1, (short)0, acc1, false, false);
    acc2 = __builtin_amdgcn_wmma_f32_16x16x32_bf16(false, afrag, false, b2, (short)0, acc2, false, false);
    acc3 = __builtin_amdgcn_wmma_f32_16x16x32_bf16(false, afrag, false, b3, (short)0, acc3, false, false);
  }

  float* cp = C + (size_t)((mt << 4) + half * 8) * N + n0 + col;
#pragma unroll
  for (int r = 0; r < 8; ++r) {
    float* rp = cp + (size_t)r * N;
    rp[0]  = acc0[r];
    rp[16] = acc1[r];
    rp[32] = acc2[r];
    rp[48] = acc3[r];
  }
}

// ---------------- attention score kernel: s = <h[n,h,:], a[h,:]> ------------

__global__ void scores_kernel(const float* __restrict__ hfeat,
                              const float* __restrict__ a_src,
                              const float* __restrict__ a_dst,
                              float* __restrict__ ssrc, float* __restrict__ sdst,
                              int Nn, int H, int C) {
  size_t t = (size_t)blockIdx.x * blockDim.x + threadIdx.x;
  if (t >= (size_t)Nn * H) return;
  int h = (int)(t % H);
  size_t n = t / H;
  const float* hp = hfeat + n * (size_t)(H * C) + (size_t)h * C;
  const float* as = a_src + (size_t)h * C;
  const float* ad = a_dst + (size_t)h * C;
  float s1 = 0.f, s2 = 0.f;
  for (int c = 0; c < C; ++c) {
    float v = hp[c];
    s1 += v * as[c];
    s2 += v * ad[c];
  }
  ssrc[t] = s1;
  sdst[t] = s2;
}

// ---------------- edge kernels ---------------------------------------------

__device__ inline void atomicMaxFloat(float* addr, float value) {
  // monotonic int/uint mapping of IEEE ordering
  if (value >= 0.f)
    atomicMax((int*)addr, __float_as_int(value));
  else
    atomicMin((unsigned int*)addr, __float_as_uint(value));
}

// e = leaky_relu(s_src[src]+s_dst[dst]); store; segment-max into mbuf[dst]
__global__ void edge_max_kernel(const int* __restrict__ ei, int E, int Nn, int H,
                                const float* __restrict__ ssrc,
                                const float* __restrict__ sdst,
                                float* __restrict__ ebuf, float* __restrict__ mbuf) {
  size_t t = (size_t)blockIdx.x * blockDim.x + threadIdx.x;
  size_t total = (size_t)(E + Nn) * H;
  if (t >= total) return;
  int h = (int)(t % H);
  size_t e = t / H;
  int src = (e < (size_t)E) ? ei[e]     : (int)(e - E);
  int dst = (e < (size_t)E) ? ei[E + e] : (int)(e - E);
  float v = ssrc[(size_t)src * H + h] + sdst[(size_t)dst * H + h];
  v = (v > 0.f) ? v : 0.2f * v;
  ebuf[t] = v;
  atomicMaxFloat(&mbuf[(size_t)dst * H + h], v);
}

// ex = exp(e - m[dst]); store; segment-sum into den[dst]
__global__ void edge_exp_kernel(const int* __restrict__ ei, int E, int Nn, int H,
                                const float* __restrict__ mbuf,
                                float* __restrict__ ebuf, float* __restrict__ den) {
  size_t t = (size_t)blockIdx.x * blockDim.x + threadIdx.x;
  size_t total = (size_t)(E + Nn) * H;
  if (t >= total) return;
  int h = (int)(t % H);
  size_t e = t / H;
  int dst = (e < (size_t)E) ? ei[E + e] : (int)(e - E);
  float ex = __expf(ebuf[t] - mbuf[(size_t)dst * H + h]);
  ebuf[t] = ex;
  atomicAdd(&den[(size_t)dst * H + h], ex);
}

// alpha = ex / (den[dst] + eps)
__global__ void edge_alpha_kernel(const int* __restrict__ ei, int E, int Nn, int H,
                                  const float* __restrict__ den, float* __restrict__ ebuf) {
  size_t t = (size_t)blockIdx.x * blockDim.x + threadIdx.x;
  size_t total = (size_t)(E + Nn) * H;
  if (t >= total) return;
  int h = (int)(t % H);
  size_t e = t / H;
  int dst = (e < (size_t)E) ? ei[E + e] : (int)(e - E);
  ebuf[t] = ebuf[t] / (den[(size_t)dst * H + h] + 1e-16f);
}

// out[dst,h,c] += alpha[e,h] * h[src,h,c]   (one thread per (edge, h*C+c))
__global__ void edge_agg_kernel(const int* __restrict__ ei, int E, int Nn,
                                int H, int C,
                                const float* __restrict__ alpha,
                                const float* __restrict__ hfeat,
                                float* __restrict__ out) {
  const int HC = H * C;
  size_t t = (size_t)blockIdx.x * blockDim.x + threadIdx.x;
  size_t total = (size_t)(E + Nn) * HC;
  if (t >= total) return;
  int c = (int)(t % HC);
  size_t e = t / HC;
  int h = c / C;
  int src = (e < (size_t)E) ? ei[e]     : (int)(e - E);
  int dst = (e < (size_t)E) ? ei[E + e] : (int)(e - E);
  float a = alpha[e * (size_t)H + h];
  atomicAdd(&out[(size_t)dst * HC + c], a * hfeat[(size_t)src * HC + c]);
}

// y = elu(agg + b), emitted as bf16 feed for next layer's WMMA GEMM
__global__ void elu_bias_bf16_kernel(const float* __restrict__ agg,
                                     const float* __restrict__ b,
                                     bf16* __restrict__ xout, int Nn, int dim) {
  size_t t = (size_t)blockIdx.x * blockDim.x + threadIdx.x;
  if (t >= (size_t)Nn * dim) return;
  int c = (int)(t % dim);
  float v = agg[t] + b[c];
  v = (v > 0.f) ? v : (__expf(v) - 1.f);
  xout[t] = (bf16)v;
}

// ---------------- layer 3: skinny GEMM (N=3) + log_softmax -----------------

__global__ void gemm3_kernel(const bf16* __restrict__ x, const float* __restrict__ W,
                             float* __restrict__ out, int Nn, int K, int NC) {
  size_t t = (size_t)blockIdx.x * blockDim.x + threadIdx.x;
  if (t >= (size_t)Nn * NC) return;
  int c = (int)(t % NC);
  size_t n = t / NC;
  const bf16* xp = x + n * (size_t)K;
  float s = 0.f;
  for (int k = 0; k < K; ++k) s += (float)xp[k] * W[(size_t)k * NC + c];
  out[t] = s;
}

__global__ void logsoftmax3_kernel(const float* __restrict__ agg,
                                   const float* __restrict__ b,
                                   float* __restrict__ out, int Nn) {
  int n = blockIdx.x * blockDim.x + threadIdx.x;
  if (n >= Nn) return;
  float v0 = agg[n * 3 + 0] + b[0];
  float v1 = agg[n * 3 + 1] + b[1];
  float v2 = agg[n * 3 + 2] + b[2];
  float m = fmaxf(v0, fmaxf(v1, v2));
  float l = m + logf(__expf(v0 - m) + __expf(v1 - m) + __expf(v2 - m));
  out[n * 3 + 0] = v0 - l;
  out[n * 3 + 1] = v1 - l;
  out[n * 3 + 2] = v2 - l;
}

// ---------------- host-side edge pipeline ----------------------------------

static void run_edges(hipStream_t stream, const int* ei, int E, int Nn, int H, int C,
                      const float* hfeat, const float* asrc, const float* adst,
                      float* ssrc, float* sdst, float* mbuf, float* den,
                      float* ebuf, float* agg) {
  size_t nh = (size_t)Nn * H;
  scores_kernel<<<cdiv_u(nh, TPB), TPB, 0, stream>>>(hfeat, asrc, adst, ssrc, sdst, Nn, H, C);
  fill_f32_kernel<<<cdiv_u(nh, TPB), TPB, 0, stream>>>(mbuf, -FLT_MAX, nh);
  fill_f32_kernel<<<cdiv_u(nh, TPB), TPB, 0, stream>>>(den, 0.f, nh);

  size_t eth = (size_t)(E + Nn) * H;
  edge_max_kernel<<<cdiv_u(eth, TPB), TPB, 0, stream>>>(ei, E, Nn, H, ssrc, sdst, ebuf, mbuf);
  edge_exp_kernel<<<cdiv_u(eth, TPB), TPB, 0, stream>>>(ei, E, Nn, H, mbuf, ebuf, den);
  edge_alpha_kernel<<<cdiv_u(eth, TPB), TPB, 0, stream>>>(ei, E, Nn, H, den, ebuf);

  size_t nd = (size_t)Nn * H * C;
  fill_f32_kernel<<<cdiv_u(nd, TPB), TPB, 0, stream>>>(agg, 0.f, nd);
  size_t etd = (size_t)(E + Nn) * H * C;
  edge_agg_kernel<<<cdiv_u(etd, TPB), TPB, 0, stream>>>(ei, E, Nn, H, C, ebuf, hfeat, agg);
}

// ---------------- entry ----------------------------------------------------

extern "C" void kernel_launch(void* const* d_in, const int* in_sizes, int n_in,
                              void* d_out, int out_size, void* d_ws, size_t ws_size,
                              hipStream_t stream) {
  const float* x   = (const float*)d_in[0];
  const int*   ei  = (const int*)  d_in[1];
  const float* W1  = (const float*)d_in[2];
  const float* as1 = (const float*)d_in[3];
  const float* ad1 = (const float*)d_in[4];
  const float* b1  = (const float*)d_in[5];
  const float* W2  = (const float*)d_in[6];
  const float* as2 = (const float*)d_in[7];
  const float* ad2 = (const float*)d_in[8];
  const float* b2  = (const float*)d_in[9];
  const float* W3  = (const float*)d_in[10];
  const float* as3 = (const float*)d_in[11];
  const float* ad3 = (const float*)d_in[12];
  const float* b3  = (const float*)d_in[13];
  float* out = (float*)d_out;

  const int F_IN = 64, HEADS = 8, HID = 128;
  const int Nn = in_sizes[0] / F_IN;     // 10000
  const int E  = in_sizes[1] / 2;        // 160000
  const int D1 = HEADS * HID;            // 1024
  const int D2 = HEADS * (HID / 2);      // 512
  const int NC = 3;

  // workspace carve-out (256B aligned slices)
  char* wp = (char*)d_ws;
  auto carve = [&](size_t bytes) -> void* {
    void* p = (void*)wp;
    wp += (bytes + 255) & ~(size_t)255;
    return p;
  };
  float* H_BUF  = (float*)carve((size_t)Nn * D1 * 4);        // GEMM output h
  float* AGG    = (float*)carve((size_t)Nn * D1 * 4);        // aggregation output
  bf16*  XBF    = (bf16*) carve((size_t)Nn * D1 * 2);        // bf16 activations
  bf16*  WBF    = (bf16*) carve((size_t)D1 * D2 * 2);        // bf16 weights (max size)
  float* SSRC   = (float*)carve((size_t)Nn * HEADS * 4);
  float* SDST   = (float*)carve((size_t)Nn * HEADS * 4);
  float* MBUF   = (float*)carve((size_t)Nn * HEADS * 4);
  float* DEN    = (float*)carve((size_t)Nn * HEADS * 4);
  float* EBUF   = (float*)carve((size_t)(E + Nn) * HEADS * 4);

  // ---------------- layer 1: 64 -> 8x128 ----------------
  {
    size_t nx = (size_t)Nn * F_IN;
    f32_to_bf16_kernel<<<cdiv_u(nx, TPB), TPB, 0, stream>>>(x, XBF, nx);
    size_t nw = (size_t)F_IN * D1;
    f32_to_bf16_kernel<<<cdiv_u(nw, TPB), TPB, 0, stream>>>(W1, WBF, nw);
    unsigned waves = (unsigned)((Nn / 16) * (D1 / 64));
    wmma_gemm_bf16_kernel<<<cdiv_u((size_t)waves, 8), TPB, 0, stream>>>(XBF, WBF, H_BUF, Nn, D1, F_IN);
    run_edges(stream, ei, E, Nn, HEADS, HID, H_BUF, as1, ad1, SSRC, SDST, MBUF, DEN, EBUF, AGG);
    elu_bias_bf16_kernel<<<cdiv_u((size_t)Nn * D1, TPB), TPB, 0, stream>>>(AGG, b1, XBF, Nn, D1);
  }

  // ---------------- layer 2: 1024 -> 8x64 ----------------
  {
    size_t nw = (size_t)D1 * D2;
    f32_to_bf16_kernel<<<cdiv_u(nw, TPB), TPB, 0, stream>>>(W2, WBF, nw);
    unsigned waves = (unsigned)((Nn / 16) * (D2 / 64));
    wmma_gemm_bf16_kernel<<<cdiv_u((size_t)waves, 8), TPB, 0, stream>>>(XBF, WBF, H_BUF, Nn, D2, D1);
    run_edges(stream, ei, E, Nn, HEADS, HID / 2, H_BUF, as2, ad2, SSRC, SDST, MBUF, DEN, EBUF, AGG);
    elu_bias_bf16_kernel<<<cdiv_u((size_t)Nn * D2, TPB), TPB, 0, stream>>>(AGG, b2, XBF, Nn, D2);
  }

  // ---------------- layer 3: 512 -> 1x3, mean(H=1) + log_softmax ----------
  {
    gemm3_kernel<<<cdiv_u((size_t)Nn * NC, TPB), TPB, 0, stream>>>(XBF, W3, H_BUF, Nn, D2, NC);
    run_edges(stream, ei, E, Nn, 1, NC, H_BUF, as3, ad3, SSRC, SDST, MBUF, DEN, EBUF, AGG);
    logsoftmax3_kernel<<<cdiv_u((size_t)Nn, TPB), TPB, 0, stream>>>(AGG, b3, out, Nn);
  }
}
